// Cluster_62294205661436
// MI455X (gfx1250) — compile-verified, matrix operand order checked by reading, not run on hardware
//
#include <hip/hip_runtime.h>
#include <hip/hip_bf16.h>

// ---------------- problem constants (match reference setup_inputs) ----------
constexpr int kB   = 64;            // batches
constexpr int kn   = 64;            // nodes per batch
constexpr int kH   = 128;           // hidden
constexpr int kNN  = kB * kn;       // 4096 nodes
constexpr int kE0  = 131072;        // B * epg explicit edges
constexpr int kET  = kE0 + kNN;     // + self loops = 135168
constexpr int kKK  = 52;            // ceil(0.8 * 64)
constexpr int kNK  = kB * kKK;      // 3328 kept nodes
constexpr float kNeg = 0.01f;       // leaky slope

typedef __attribute__((ext_vector_type(16))) __bf16 bf16x16;
typedef __attribute__((ext_vector_type(4)))  __bf16 bf16x4;
typedef __attribute__((ext_vector_type(8)))  float  f32x8;

__device__ __forceinline__ float leaky(float v) { return v > 0.f ? v : kNeg * v; }

// K index inside a 16x32 bf16 WMMA fragment (ISA 7.12.2): element i of the
// 16-element-per-lane fragment maps to k = (i/8)*16 + (lane/16)*8 + (i%8).
// Consecutive i within each half -> consecutive k, so [row][k]-layout LDS
// panels give two wide (8-elem) contiguous loads per fragment.
__device__ __forceinline__ int frag_k(int i, int lane) {
  return ((i >> 3) << 4) + ((lane >> 4) << 3) + (i & 7);
}

__device__ __forceinline__ bf16x4 cvt4(float4 f) {
  bf16x4 r;
  r[0] = (__bf16)f.x; r[1] = (__bf16)f.y; r[2] = (__bf16)f.z; r[3] = (__bf16)f.w;
  return r;
}

__device__ __forceinline__ void edge_rc(int e, const int* __restrict__ ei,
                                        int& r, int& c) {
  if (e < kE0) { r = ei[e]; c = ei[kE0 + e]; }
  else         { r = e - kE0; c = r; }
}

// ---------------- LDS-tiled WMMA bf16 GEMM -----------------------------------
// C[MxN] = act(A[MxK] @ B[KxN]); A,B,C f32 row-major, staged into LDS as bf16
// (converted once per panel). Block = 256 threads = 8 waves, block tile
// 128x128, wave tile 32x64 (2x4 accumulators -> 8 v_wmma per 32-wide K-panel).
// Both panels use [row][k] layout with padded pitch 36 (72B: 18-dword row
// stride -> conflict-free fragment loads; k-contiguous -> wide ds loads).
// All problem dims are multiples of the tile: no guards, EXEC stays all-ones.
constexpr int BM = 128, BN = 128, BK = 32;
constexpr int KP = BK + 4;          // padded k pitch (bf16 elems)

template<bool TRANSA, bool LEAKY>
__global__ __launch_bounds__(256)
void wmma_gemm_kernel(const float* __restrict__ A,
                      const float* __restrict__ Bm,
                      float* __restrict__ C,
                      int M, int Nc, int K,
                      int lda, int ldb, int ldc) {
  __shared__ __bf16 As[BM * KP];    // A panel, [m][k]
  __shared__ __bf16 Bs[BN * KP];    // B panel transposed, [n][k]

  const int lane = threadIdx.x & 31;
  const int wave = threadIdx.x >> 5;
  const int tn0 = blockIdx.x * BN;
  const int tm0 = blockIdx.y * BM;
  const int wm = (wave >> 1) * 32;       // wave row offset inside block tile
  const int wn = (wave & 1) * 64;        // wave col offset inside block tile

  f32x8 acc[2][4] = {};

  for (int k0 = 0; k0 < K; k0 += BK) {
    __syncthreads();                     // prior frag reads done before refill
#pragma unroll
    for (int v = 0; v < 4; ++v) {
      const int chunk = threadIdx.x + v * 256;     // 1024 chunks of 4 floats
      // ---- stage A panel ----
      if (!TRANSA) {                                // contiguous along k
        const int m = chunk >> 3, kc = (chunk & 7) << 2;
        float4 f = *(const float4*)&A[(size_t)(tm0 + m) * lda + k0 + kc];
        *(bf16x4*)&As[m * KP + kc] = cvt4(f);
        if (k0 + BK < K)
          __builtin_prefetch(&A[(size_t)(tm0 + m) * lda + k0 + BK + kc], 0, 1);
      } else {                                      // contiguous along m
        const int k = chunk >> 5, mc = (chunk & 31) << 2;
        float4 f = *(const float4*)&A[(size_t)(k0 + k) * lda + tm0 + mc];
        As[(mc + 0) * KP + k] = (__bf16)f.x;        // transpose on store
        As[(mc + 1) * KP + k] = (__bf16)f.y;
        As[(mc + 2) * KP + k] = (__bf16)f.z;
        As[(mc + 3) * KP + k] = (__bf16)f.w;
        if (k0 + BK < K)
          __builtin_prefetch(&A[(size_t)(k0 + BK + k) * lda + tm0 + mc], 0, 1);
      }
      // ---- stage B panel (contiguous along n, transpose on store) ----
      {
        const int k = chunk >> 5, nc = (chunk & 31) << 2;
        float4 g = *(const float4*)&Bm[(size_t)(k0 + k) * ldb + tn0 + nc];
        Bs[(nc + 0) * KP + k] = (__bf16)g.x;
        Bs[(nc + 1) * KP + k] = (__bf16)g.y;
        Bs[(nc + 2) * KP + k] = (__bf16)g.z;
        Bs[(nc + 3) * KP + k] = (__bf16)g.w;
        if (k0 + BK < K)
          __builtin_prefetch(&Bm[(size_t)(k0 + BK + k) * ldb + tn0 + nc], 0, 1);
      }
    }
    __syncthreads();

    // ---- load fragments from LDS (k-contiguous -> wide ds loads), 8 WMMAs --
    bf16x16 af[2], bfr[4];
#pragma unroll
    for (int ii = 0; ii < 2; ++ii) {
      const int m = wm + ii * 16 + (lane & 15);
#pragma unroll
      for (int e = 0; e < 16; ++e)
        af[ii][e] = As[m * KP + frag_k(e, lane)];
    }
#pragma unroll
    for (int jj = 0; jj < 4; ++jj) {
      const int nn = wn + jj * 16 + (lane & 15);
#pragma unroll
      for (int e = 0; e < 16; ++e)
        bfr[jj][e] = Bs[nn * KP + frag_k(e, lane)];
    }
#pragma unroll
    for (int ii = 0; ii < 2; ++ii)
#pragma unroll
      for (int jj = 0; jj < 4; ++jj)
        acc[ii][jj] = __builtin_amdgcn_wmma_f32_16x16x32_bf16(
            false, af[ii], false, bfr[jj], (short)0, acc[ii][jj], false, false);
  }

  // ---- epilogue: C/D layout M = r + 8*(lane>=16), N = lane&15 ----
  const int rbase = (lane >> 4) << 3;
#pragma unroll
  for (int ii = 0; ii < 2; ++ii)
#pragma unroll
    for (int jj = 0; jj < 4; ++jj) {
      const int col = tn0 + wn + jj * 16 + (lane & 15);
#pragma unroll
      for (int r = 0; r < 8; ++r) {
        const int row = tm0 + wm + ii * 16 + rbase + r;
        float v = acc[ii][jj][r];
        if (LEAKY) v = leaky(v);
        C[(size_t)row * ldc + col] = v;
      }
    }
}

// ---------------- graph normalization ---------------------------------------
__global__ void deg_kernel(const int* __restrict__ ei, const float* __restrict__ w_in,
                           float* __restrict__ deg) {
  int e = blockIdx.x * blockDim.x + threadIdx.x;
  if (e >= kET) return;
  int r, c; edge_rc(e, ei, r, c);
  float w = (e < kE0) ? w_in[e] : 1.f;
  atomicAdd(&deg[c], w);
}

__global__ void dinv_kernel(const float* __restrict__ deg, float* __restrict__ dinv) {
  int i = blockIdx.x * blockDim.x + threadIdx.x;
  if (i >= kNN) return;
  float d = deg[i];
  dinv[i] = d > 0.f ? rsqrtf(fmaxf(d, 1e-12f)) : 0.f;
}

__global__ void ew_kernel(const int* __restrict__ ei, const float* __restrict__ w_in,
                          const float* __restrict__ dinv, float* __restrict__ ew) {
  int e = blockIdx.x * blockDim.x + threadIdx.x;
  if (e >= kET) return;
  int r, c; edge_rc(e, ei, r, c);
  float w = (e < kE0) ? w_in[e] : 1.f;
  ew[e] = dinv[r] * w * dinv[c];
}

// SpMM: Y[c,:] += w[e] * X[r,:]  (one block of kH threads per edge)
__global__ void spmm_kernel(const int* __restrict__ ei, const float* __restrict__ w,
                            const float* __restrict__ X, float* __restrict__ Y) {
  int e = blockIdx.x, f = threadIdx.x;
  int r, c; edge_rc(e, ei, r, c);
  atomicAdd(&Y[(size_t)c * kH + f], X[(size_t)r * kH + f] * w[e]);
}

// ---------------- attention pieces -------------------------------------------
__global__ void qt_kernel(const float* __restrict__ tx, float* __restrict__ qt) {
  int i = blockIdx.x, f = threadIdx.x;
  qt[(size_t)i * kH + f] = tx[(size_t)(i >> 6) * kH + f];
}

__global__ void build_h_kernel(const float* __restrict__ a, const float* __restrict__ q,
                               float* __restrict__ h) {
  size_t idx = (size_t)blockIdx.x * blockDim.x + threadIdx.x;
  if (idx >= (size_t)kNN * 4 * kH) return;
  int j = (int)(idx % (4 * kH));
  int i = (int)(idx / (4 * kH));
  int seg = j / kH, jj = j % kH;
  float av = a[(size_t)i * kH + jj], qv = q[(size_t)i * kH + jj];
  h[idx] = seg == 0 ? av : seg == 1 ? qv : seg == 2 ? (av - qv) : (av * qv);
}

// per-row dot with W3 (Hx1) + leaky; one wave per row
__global__ void rowdot_kernel(const float* __restrict__ h2, const float* __restrict__ w3,
                              float* __restrict__ out) {
  int i = blockIdx.x, lane = threadIdx.x;
  float s = 0.f;
#pragma unroll
  for (int t = 0; t < kH; t += 32) s += h2[(size_t)i * kH + lane + t] * w3[lane + t];
  for (int o = 16; o > 0; o >>= 1) s += __shfl_xor(s, o, 32);
  if (lane == 0) out[i] = leaky(s);
}

// softmax over each contiguous batch segment of 64 rows
__global__ void softmax_batch_kernel(const float* __restrict__ in, float* __restrict__ out) {
  __shared__ float sh[64];
  int b = blockIdx.x, t = threadIdx.x;
  float v = in[b * 64 + t];
  sh[t] = v; __syncthreads();
  for (int o = 32; o > 0; o >>= 1) { if (t < o) sh[t] = fmaxf(sh[t], sh[t + o]); __syncthreads(); }
  float m = sh[0]; __syncthreads();
  float e = __expf(v - m);
  sh[t] = e; __syncthreads();
  for (int o = 32; o > 0; o >>= 1) { if (t < o) sh[t] += sh[t + o]; __syncthreads(); }
  out[b * 64 + t] = e / sh[0];
}

// ---------------- edge softmax over destination column -----------------------
__global__ void edge_logit_kernel(const int* __restrict__ ei, const float* __restrict__ f1,
                                  const float* __restrict__ f2, float* __restrict__ E,
                                  float* __restrict__ colmax) {
  int e = blockIdx.x * blockDim.x + threadIdx.x;
  if (e >= kET) return;
  int r, c; edge_rc(e, ei, r, c);
  float v = leaky(f1[c] + f2[r]);          // softmax outputs -> v > 0 always
  E[e] = v;
  atomicMax((int*)&colmax[c], __float_as_int(v));  // valid for positive floats
}

__global__ void edge_exp_kernel(const int* __restrict__ ei, float* __restrict__ E,
                                const float* __restrict__ colmax, float* __restrict__ colsum) {
  int e = blockIdx.x * blockDim.x + threadIdx.x;
  if (e >= kET) return;
  int r, c; edge_rc(e, ei, r, c);
  float ee = __expf(E[e] - colmax[c]);
  E[e] = ee;
  atomicAdd(&colsum[c], ee);
}

__global__ void edge_norm_kernel(const int* __restrict__ ei, float* __restrict__ E,
                                 const float* __restrict__ colsum) {
  int e = blockIdx.x * blockDim.x + threadIdx.x;
  if (e >= kET) return;
  int r, c; edge_rc(e, ei, r, c);
  E[e] /= colsum[c];
}

// ---------------- head combine, score, top-k ---------------------------------
__global__ void head_combine_kernel(const float* __restrict__ x, const float* __restrict__ G,
                                    float* __restrict__ xc, int mode) {
  int i = blockIdx.x * blockDim.x + threadIdx.x;
  if (i >= kNN * kH) return;
  float v = leaky(x[i] + G[i]);
  xc[i] = (mode == 0) ? v : (xc[i] + v) * 0.5f;   // average over HEADS=2
}

__global__ void add_kernel(const float* __restrict__ a, const float* __restrict__ b,
                           float* __restrict__ o) {
  int i = blockIdx.x * blockDim.x + threadIdx.x;
  if (i >= kNN) return;
  o[i] = a[i] + b[i];
}

// stable descending argsort rank; matches jax.lax.top_k tie-breaking
__global__ void topk_kernel(const float* __restrict__ score, int* __restrict__ perm) {
  __shared__ float sh[64];
  int b = blockIdx.x, t = threadIdx.x;
  float v = score[b * 64 + t];
  sh[t] = v; __syncthreads();
  int rank = 0;
  for (int j = 0; j < 64; ++j) {
    float vj = sh[j];
    rank += (vj > v) || (vj == v && j < t);
  }
  if (rank < kKK) perm[b * kKK + rank] = b * 64 + t;
}

__global__ void xout_kernel(const float* __restrict__ x, const float* __restrict__ score,
                            const int* __restrict__ perm, float* __restrict__ xout,
                            float* __restrict__ bout, float* __restrict__ pout) {
  int i = blockIdx.x, f = threadIdx.x;
  int p = perm[i];
  xout[(size_t)i * kH + f] = x[(size_t)p * kH + f] * score[p];
  if (f == 0) { bout[i] = (float)(p >> 6); pout[i] = (float)p; }
}

// ---------------- dense A_d / S_p assembly -----------------------------------
__global__ void pos_kernel(const int* __restrict__ perm, int* __restrict__ pos) {
  int j = blockIdx.x * blockDim.x + threadIdx.x;
  if (j >= kNK) return;
  pos[perm[j]] = j;
}

__global__ void scatter_A_kernel(const int* __restrict__ ei, const float* __restrict__ ew,
                                 float* __restrict__ Ad) {
  int e = blockIdx.x * blockDim.x + threadIdx.x;
  if (e >= kET) return;
  int r, c; edge_rc(e, ei, r, c);
  atomicAdd(&Ad[(size_t)r * kNN + c], ew[e]);
}

__global__ void scatter_S_kernel(const int* __restrict__ ei, const float* __restrict__ E,
                                 const int* __restrict__ pos, float* __restrict__ Sp) {
  int e = blockIdx.x * blockDim.x + threadIdx.x;
  if (e >= kET) return;
  int r, c; edge_rc(e, ei, r, c);
  int j = pos[c];
  if (j >= 0) atomicAdd(&Sp[(size_t)r * kNK + j], E[e]);  // E_g == E (both heads)
}

__global__ void diag_kernel(float* __restrict__ A2) {
  int i = blockIdx.x * blockDim.x + threadIdx.x;
  if (i >= kNK) return;
  A2[(size_t)i * kNK + i] = 1.0f;
}

// ---------------- host orchestration -----------------------------------------
extern "C" void kernel_launch(void* const* d_in, const int* in_sizes, int n_in,
                              void* d_out, int out_size, void* d_ws, size_t ws_size,
                              hipStream_t stream) {
  const float* x   = (const float*)d_in[0];
  const int*   ei  = (const int*)  d_in[1];
  const float* wIn = (const float*)d_in[2];
  const float* tx  = (const float*)d_in[3];
  const int*   bix = (const int*)  d_in[4]; (void)bix; (void)in_sizes; (void)n_in; (void)out_size; (void)ws_size;
  const float* Wk  = (const float*)d_in[5];
  const float* W1  = (const float*)d_in[6];
  const float* W2  = (const float*)d_in[7];
  const float* W3  = (const float*)d_in[8];
  const float* Lw  = (const float*)d_in[9];

  // workspace carve (~205 MB)
  char* wp = (char*)d_ws;
  auto alloc = [&](size_t bytes) -> void* {
    void* p = (void*)wp; wp += (bytes + 255) & ~(size_t)255; return p;
  };
  float* deg    = (float*)alloc((size_t)kNN * 4);
  float* dinv   = (float*)alloc((size_t)kNN * 4);
  float* ew     = (float*)alloc((size_t)kET * 4);
  float* Ebuf   = (float*)alloc((size_t)kET * 4);
  float* colmax = (float*)alloc((size_t)kNN * 4);
  float* colsum = (float*)alloc((size_t)kNN * 4);
  float* qt     = (float*)alloc((size_t)kNN * kH * 4);
  float* xq     = (float*)alloc((size_t)kNN * kH * 4);
  float* xtmp   = (float*)alloc((size_t)kNN * kH * 4);
  float* abuf   = (float*)alloc((size_t)kNN * kH * 4);
  float* hbuf   = (float*)alloc((size_t)kNN * 4 * kH * 4);
  float* h1     = (float*)alloc((size_t)kNN * 2 * kH * 4);
  float* h2     = (float*)alloc((size_t)kNN * kH * 4);
  float* rowtmp = (float*)alloc((size_t)kNN * 4);
  float* fv0    = (float*)alloc((size_t)kNN * 4);
  float* fv1    = (float*)alloc((size_t)kNN * 4);
  float* fv2    = (float*)alloc((size_t)kNN * 4);
  float* fv3    = (float*)alloc((size_t)kNN * 4);
  float* score  = (float*)alloc((size_t)kNN * 4);
  float* agg    = (float*)alloc((size_t)kNN * kH * 4);
  float* Gh     = (float*)alloc((size_t)kNN * kH * 4);
  float* xc     = (float*)alloc((size_t)kNN * kH * 4);
  int*   pos    = (int*)  alloc((size_t)kNN * 4);
  int*   perm   = (int*)  alloc((size_t)kNK * 4);
  float* Ad     = (float*)alloc((size_t)kNN * kNN * 4);
  float* Sp     = (float*)alloc((size_t)kNN * kNK * 4);
  float* T      = (float*)alloc((size_t)kNN * kNK * 4);

  float* xout = (float*)d_out;
  float* A2   = xout + (size_t)kNK * kH;
  float* bout = A2 + (size_t)kNK * kNK;
  float* pout = bout + kNK;

  const int TB = 256;
  const int gE = (kET + TB - 1) / TB;
  const int gN = (kNN + TB - 1) / TB;

  auto gemm = [&](const float* A, const float* Bm, float* C,
                  int M, int Nc, int K, bool lk) {
    dim3 g(Nc / BN, M / BM);               // all M,Nc here are multiples of 128
    if (lk) wmma_gemm_kernel<false, true ><<<g, 256, 0, stream>>>(A, Bm, C, M, Nc, K, K, Nc, Nc);
    else    wmma_gemm_kernel<false, false><<<g, 256, 0, stream>>>(A, Bm, C, M, Nc, K, K, Nc, Nc);
  };

  auto attention = [&](const float* kv, const float* q, int wi, float* fout) {
    gemm(kv, Wk + (size_t)wi * kH * kH, abuf, kNN, kH, kH, false);           // a = kv@Wk
    build_h_kernel<<<(kNN * 4 * kH + TB - 1) / TB, TB, 0, stream>>>(abuf, q, hbuf);
    gemm(hbuf, W1 + (size_t)wi * 4 * kH * 2 * kH, h1, kNN, 2 * kH, 4 * kH, true);
    gemm(h1,   W2 + (size_t)wi * 2 * kH * kH,     h2, kNN, kH, 2 * kH, true);
    rowdot_kernel<<<kNN, 32, 0, stream>>>(h2, W3 + (size_t)wi * kH, rowtmp);
    softmax_batch_kernel<<<kB, 64, 0, stream>>>(rowtmp, fout);
  };

  // 1) symmetric normalization
  hipMemsetAsync(deg, 0, (size_t)kNN * 4, stream);
  deg_kernel<<<gE, TB, 0, stream>>>(ei, wIn, deg);
  dinv_kernel<<<gN, TB, 0, stream>>>(deg, dinv);
  ew_kernel<<<gE, TB, 0, stream>>>(ei, wIn, dinv, ew);

  // 2) x_q = hop(hop(x))
  hipMemsetAsync(xtmp, 0, (size_t)kNN * kH * 4, stream);
  spmm_kernel<<<kET, kH, 0, stream>>>(ei, ew, x, xtmp);
  hipMemsetAsync(xq, 0, (size_t)kNN * kH * 4, stream);
  spmm_kernel<<<kET, kH, 0, stream>>>(ei, ew, xtmp, xq);

  // 3) qt broadcast, then f1 / f2
  qt_kernel<<<kNN, kH, 0, stream>>>(tx, qt);
  attention(x, xq, 0, fv0);
  attention(x, qt, 1, fv1);

  // 4) edge softmax over columns
  hipMemsetAsync(colmax, 0, (size_t)kNN * 4, stream);   // logits > 0, so 0-init max is safe
  hipMemsetAsync(colsum, 0, (size_t)kNN * 4, stream);
  edge_logit_kernel<<<gE, TB, 0, stream>>>(ei, fv0, fv1, Ebuf, colmax);
  edge_exp_kernel  <<<gE, TB, 0, stream>>>(ei, Ebuf, colmax, colsum);
  edge_norm_kernel <<<gE, TB, 0, stream>>>(ei, Ebuf, colsum);

  // 5) agg = segsum(x[row]*E, col); x_c = mean_h leaky(x + agg @ lin_W[h])
  hipMemsetAsync(agg, 0, (size_t)kNN * kH * 4, stream);
  spmm_kernel<<<kET, kH, 0, stream>>>(ei, Ebuf, x, agg);
  gemm(agg, Lw,                       Gh, kNN, kH, kH, false);
  head_combine_kernel<<<(kNN * kH + TB - 1) / TB, TB, 0, stream>>>(x, Gh, xc, 0);
  gemm(agg, Lw + (size_t)kH * kH,     Gh, kNN, kH, kH, false);
  head_combine_kernel<<<(kNN * kH + TB - 1) / TB, TB, 0, stream>>>(x, Gh, xc, 1);

  // 6) x_q2 = hop(hop(x_c)); g1 / g2; cluster score; top-k
  hipMemsetAsync(xtmp, 0, (size_t)kNN * kH * 4, stream);
  spmm_kernel<<<kET, kH, 0, stream>>>(ei, ew, xc, xtmp);
  hipMemsetAsync(xq, 0, (size_t)kNN * kH * 4, stream);
  spmm_kernel<<<kET, kH, 0, stream>>>(ei, ew, xtmp, xq);
  attention(xc, xq, 2, fv2);
  attention(xc, qt, 3, fv3);
  add_kernel<<<gN, TB, 0, stream>>>(fv2, fv3, rowtmp);
  softmax_batch_kernel<<<kB, 64, 0, stream>>>(rowtmp, score);
  topk_kernel<<<kB, 64, 0, stream>>>(score, perm);
  xout_kernel<<<kNK, kH, 0, stream>>>(x, score, perm, xout, bout, pout);

  // 7) dense A_d, S_p; A2 = S_p^T @ (A_d @ S_p) via bf16 WMMA (L2-resident)
  hipMemsetAsync(Ad, 0, (size_t)kNN * kNN * 4, stream);
  scatter_A_kernel<<<gE, TB, 0, stream>>>(ei, ew, Ad);
  hipMemsetAsync(pos, 0xFF, (size_t)kNN * 4, stream);   // -1
  pos_kernel<<<(kNK + TB - 1) / TB, TB, 0, stream>>>(perm, pos);
  hipMemsetAsync(Sp, 0, (size_t)kNN * kNK * 4, stream);
  scatter_S_kernel<<<gE, TB, 0, stream>>>(ei, Ebuf, pos, Sp);

  {
    dim3 g1(kNK / BN, kNN / BM);   // T = Ad @ Sp : 4096x3328, K=4096
    wmma_gemm_kernel<false, false><<<g1, 256, 0, stream>>>(Ad, Sp, T, kNN, kNK, kNN, kNN, kNK, kNK);
    dim3 g2(kNK / BN, kNK / BM);   // A2 = Sp^T @ T : 3328x3328, K=4096
    wmma_gemm_kernel<true, false><<<g2, 256, 0, stream>>>(Sp, T, A2, kNK, kNK, kNN, kNK, kNK, kNK);
  }
  diag_kernel<<<(kNK + TB - 1) / TB, TB, 0, stream>>>(A2);
}
